// BitLinear_70360154243396
// MI455X (gfx1250) — matrix operand order repaired and checked by reading, not compiled, not run on hardware
//
#include <hip/hip_runtime.h>

#define M_TOT 16384   // B*S
#define K_TOT 2048    // DIN
#define N_TOT 8192    // DOUT

#define BM 128
#define BN 128
#define BK 32
#define LDST 40       // padded LDS row stride (halfs): 80B, 16B-aligned rows

typedef _Float16 v16h __attribute__((ext_vector_type(16)));
typedef _Float16 v8h  __attribute__((ext_vector_type(8)));
typedef _Float16 v4h  __attribute__((ext_vector_type(4)));
typedef float    v8f  __attribute__((ext_vector_type(8)));
typedef unsigned int u32x4 __attribute__((ext_vector_type(4)));
typedef int          i32x8 __attribute__((ext_vector_type(8)));
typedef int          i32x4 __attribute__((ext_vector_type(4)));

#define HAVE_TDM __has_builtin(__builtin_amdgcn_tensor_load_to_lds)

// ---- CDNA5 async global->LDS copy (ASYNCcnt-tracked), 16B per lane ----
__device__ __forceinline__ void async_g2l_b128(unsigned lds_off, const void* gptr) {
  unsigned long long ga = (unsigned long long)(uintptr_t)gptr;
  asm volatile("global_load_async_to_lds_b128 %0, %1, off"
               :: "v"(lds_off), "v"(ga) : "memory");
}
__device__ __forceinline__ void wait_async_le4() {
#if __has_builtin(__builtin_amdgcn_s_wait_asynccnt)
  __builtin_amdgcn_s_wait_asynccnt(4);
#else
  asm volatile("s_wait_asynccnt 0x4" ::: "memory");
#endif
}
__device__ __forceinline__ void wait_async0() {
#if __has_builtin(__builtin_amdgcn_s_wait_asynccnt)
  __builtin_amdgcn_s_wait_asynccnt(0);
#else
  asm volatile("s_wait_asynccnt 0x0" ::: "memory");
#endif
}
__device__ __forceinline__ void wait_tensor_le2() {
#if __has_builtin(__builtin_amdgcn_s_wait_tensorcnt)
  __builtin_amdgcn_s_wait_tensorcnt(2);
#else
  asm volatile("s_wait_tensorcnt 0x2" ::: "memory");
#endif
}
__device__ __forceinline__ void wait_tensor0() {
#if __has_builtin(__builtin_amdgcn_s_wait_tensorcnt)
  __builtin_amdgcn_s_wait_tensorcnt(0);
#else
  asm volatile("s_wait_tensorcnt 0x0" ::: "memory");
#endif
}

#if HAVE_TDM
// D# group1 for a 2D tile (BM rows x BK halfs) of a (rows x K_TOT) f16 tensor,
// row stride K_TOT; LDS pad 4 DWORDs after every 16 DWORDs -> 80B row stride.
__device__ __forceinline__ i32x8 tdm_g1(unsigned rows_remaining) {
  i32x8 g1;
  g1[0] = (1 << 16) | (1 << 20) | (3 << 22) | (3 << 25);  // 2B elems, pad 4DW/16DW
  g1[1] = (int)((K_TOT & 0xffff) << 16);                  // tensor_dim0[15:0]
  g1[2] = (int)(((K_TOT >> 16) & 0xffff) |
                ((rows_remaining & 0xffff) << 16));       // dim0[31:16] | dim1[15:0]
  g1[3] = (int)(((rows_remaining >> 16) & 0xffff) |
                ((unsigned)BK << 16));                    // dim1[31:16] | tile_dim0
  g1[4] = (int)BM;                                        // tile_dim1 | tile_dim2=0
  g1[5] = (int)K_TOT;                                     // tensor_dim0_stride[31:0]
  g1[6] = 0;
  g1[7] = 0;
  return g1;
}
__device__ __forceinline__ void tdm_issue(const i32x8& g1, unsigned lds_off,
                                          const void* gbase) {
  unsigned long long ga = (unsigned long long)(uintptr_t)gbase;
  u32x4 g0;
  g0[0] = 1u;                                                 // count=1, user mode
  g0[1] = lds_off;                                            // lds_addr (bytes)
  g0[2] = (unsigned)(ga & 0xffffffffu);                       // global_addr[31:0]
  g0[3] = (unsigned)((ga >> 32) & 0x01ffffffu) | (2u << 30);  // addr[56:32], type=2
  i32x4 z4 = {0, 0, 0, 0};
  i32x8 z8 = {0, 0, 0, 0, 0, 0, 0, 0};
  __builtin_amdgcn_tensor_load_to_lds(g0, g1, z4, z4, z8, 0);
}
#endif

// ---------------- pass 1: partial sums of |w| ----------------
__global__ __launch_bounds__(256) void absw_partial(const float* __restrict__ w,
                                                    float* __restrict__ partials) {
  __shared__ float red[256];
  const float4* w4 = (const float4*)w;
  size_t base = (size_t)blockIdx.x * 4096;
  float s = 0.f;
#pragma unroll
  for (int i = 0; i < 16; ++i) {
    float4 v = w4[base + threadIdx.x + i * 256];
    s += fabsf(v.x) + fabsf(v.y) + fabsf(v.z) + fabsf(v.w);
  }
  red[threadIdx.x] = s;
  __syncthreads();
  for (int off = 128; off > 0; off >>= 1) {
    if (threadIdx.x < (unsigned)off) red[threadIdx.x] += red[threadIdx.x + off];
    __syncthreads();
  }
  if (threadIdx.x == 0) partials[blockIdx.x] = red[0];
}

// ---------------- pass 2: finalize threshold ----------------
__global__ __launch_bounds__(256) void absw_final(const float* __restrict__ partials,
                                                  float* __restrict__ thr) {
  __shared__ float red[256];
  float s = 0.f;
#pragma unroll
  for (int i = 0; i < 4; ++i) s += partials[threadIdx.x + i * 256];
  red[threadIdx.x] = s;
  __syncthreads();
  for (int off = 128; off > 0; off >>= 1) {
    if (threadIdx.x < (unsigned)off) red[threadIdx.x] += red[threadIdx.x + off];
    __syncthreads();
  }
  if (threadIdx.x == 0) thr[0] = red[0] * (1.0f / (float)((size_t)N_TOT * K_TOT));
}

// ---------------- ternarize weights -> f16 ----------------
__global__ __launch_bounds__(256) void quant_w(const float* __restrict__ w,
                                               const float* __restrict__ thrp,
                                               _Float16* __restrict__ wh) {
  const float thr = thrp[0];
  const float4* w4 = (const float4*)w;
  v4h* o4 = (v4h*)wh;
  size_t base = (size_t)blockIdx.x * 1024 + threadIdx.x;
#pragma unroll
  for (int i = 0; i < 4; ++i) {
    size_t idx = base + (size_t)i * 256;
    float4 v = w4[idx];
    v4h o;
    o[0] = (_Float16)((fabsf(v.x) > thr) ? (v.x > 0.f ? 1.f : -1.f) : 0.f);
    o[1] = (_Float16)((fabsf(v.y) > thr) ? (v.y > 0.f ? 1.f : -1.f) : 0.f);
    o[2] = (_Float16)((fabsf(v.z) > thr) ? (v.z > 0.f ? 1.f : -1.f) : 0.f);
    o[3] = (_Float16)((fabsf(v.w) > thr) ? (v.w > 0.f ? 1.f : -1.f) : 0.f);
    o4[idx] = o;
  }
}

// ---------------- RMSNorm rows -> f16 ----------------
__global__ __launch_bounds__(256) void rmsnorm_f16(const float* __restrict__ x,
                                                   const float* __restrict__ gamma,
                                                   _Float16* __restrict__ xh) {
  __shared__ float red[256];
  const int row = blockIdx.x;
  const int tid = threadIdx.x;
  const float4* x4 = (const float4*)(x + (size_t)row * K_TOT);
  const float4* g4 = (const float4*)gamma;
  float4 v0 = x4[tid];
  float4 v1 = x4[tid + 256];
  float s = v0.x * v0.x + v0.y * v0.y + v0.z * v0.z + v0.w * v0.w +
            v1.x * v1.x + v1.y * v1.y + v1.z * v1.z + v1.w * v1.w;
  red[tid] = s;
  __syncthreads();
  for (int off = 128; off > 0; off >>= 1) {
    if (tid < off) red[tid] += red[tid + off];
    __syncthreads();
  }
  const float ms = red[0] * (1.0f / (float)K_TOT);
  const float sc = rsqrtf(ms + 1.1920928955078125e-7f);
  float4 g0 = g4[tid];
  float4 g1 = g4[tid + 256];
  v4h o0, o1;
  o0[0] = (_Float16)(v0.x * sc * g0.x);
  o0[1] = (_Float16)(v0.y * sc * g0.y);
  o0[2] = (_Float16)(v0.z * sc * g0.z);
  o0[3] = (_Float16)(v0.w * sc * g0.w);
  o1[0] = (_Float16)(v1.x * sc * g1.x);
  o1[1] = (_Float16)(v1.y * sc * g1.y);
  o1[2] = (_Float16)(v1.z * sc * g1.z);
  o1[3] = (_Float16)(v1.w * sc * g1.w);
  v4h* out4 = (v4h*)(xh + (size_t)row * K_TOT);
  out4[tid] = o0;
  out4[tid + 256] = o1;
}

// ---------------- WMMA GEMM: out[m,n] = sum_k xh[m,k]*wh[n,k] + bias[n] ----------------
__global__ __launch_bounds__(256, 1) void gemm_wmma(const _Float16* __restrict__ A,
                                                    const _Float16* __restrict__ W,
                                                    const float* __restrict__ bias,
                                                    float* __restrict__ C) {
  __shared__ _Float16 lds[3][2][BM * LDST];  // 3 bufs x (A,W), 61440 bytes

  const int tid   = threadIdx.x;
  const int lane  = tid & 31;
  const int wave  = tid >> 5;
  const int waveM = wave & 1;    // 2 waves along M (64 rows each)
  const int waveN = wave >> 1;   // 4 waves along N (32 cols each)
  const int lh    = lane & 15;
  const int hh    = lane >> 4;
  const int m0    = blockIdx.y * BM;
  const int n0    = blockIdx.x * BN;

  const _Float16* Ag = A + (size_t)m0 * K_TOT;
  const _Float16* Wg = W + (size_t)n0 * K_TOT;

  v8f acc[4][2];
#pragma unroll
  for (int i = 0; i < 4; ++i)
#pragma unroll
    for (int j = 0; j < 2; ++j)
#pragma unroll
      for (int r = 0; r < 8; ++r) acc[i][j][r] = 0.f;

#if HAVE_TDM
  const i32x8 g1A = tdm_g1((unsigned)(M_TOT - m0));
  const i32x8 g1W = tdm_g1((unsigned)(N_TOT - n0));
  auto stage_load = [&](int ks, int b) {
    if (wave == 0) {
      const size_t kbytes = (size_t)ks * (BK * 2);  // 64B per stage
      tdm_issue(g1A, (unsigned)(uintptr_t)&lds[b][0][0], (const char*)Ag + kbytes);
      tdm_issue(g1W, (unsigned)(uintptr_t)&lds[b][1][0], (const char*)Wg + kbytes);
    }
  };
  auto wait_steady = [&]() { if (wave == 0) wait_tensor_le2(); };
  auto wait_last   = [&]() { if (wave == 0) wait_tensor0(); };
#else
  auto stage_load = [&](int ks, int b) {
    const size_t kof = (size_t)ks * BK;
#pragma unroll
    for (int i = 0; i < 2; ++i) {
      const int c = tid + i * 256;
      const int row = c >> 2, kc = c & 3;
      unsigned la = (unsigned)(uintptr_t)&lds[b][0][row * LDST + kc * 8];
      unsigned lb = (unsigned)(uintptr_t)&lds[b][1][row * LDST + kc * 8];
      async_g2l_b128(la, Ag + (size_t)row * K_TOT + kof + kc * 8);
      async_g2l_b128(lb, Wg + (size_t)row * K_TOT + kof + kc * 8);
    }
  };
  auto wait_steady = [&]() { wait_async_le4(); };
  auto wait_last   = [&]() { wait_async0(); };
#endif

  auto compute = [&](int b) {
    const _Float16* lA = &lds[b][0][0];
    const _Float16* lW = &lds[b][1][0];
    v16h af[4], bf[2];
#pragma unroll
    for (int i = 0; i < 4; ++i) {
      // A 16x32 f16 layout: lane m -> K0..7 (v0-3) & K16..23 (v4-7); lane m+16 -> +8
      const v8h* p = (const v8h*)&lA[(waveM * 64 + i * 16 + lh) * LDST + hh * 8];
      union { v16h v; v8h h[2]; } u;
      u.h[0] = p[0];
      u.h[1] = p[2];  // +16 halfs
      af[i] = u.v;
    }
#pragma unroll
    for (int j = 0; j < 2; ++j) {
      // B 32x16 f16 layout: lane n -> K0..7 (v0-3) & K16..23 (v4-7); lane n+16 -> +8
      const v8h* p = (const v8h*)&lW[(waveN * 32 + j * 16 + lh) * LDST + hh * 8];
      union { v16h v; v8h h[2]; } u;
      u.h[0] = p[0];
      u.h[1] = p[2];
      bf[j] = u.v;
    }
#pragma unroll
    for (int i = 0; i < 4; ++i)
#pragma unroll
      for (int j = 0; j < 2; ++j)
        acc[i][j] = __builtin_amdgcn_wmma_f32_16x16x32_f16(
            false, af[i], false, bf[j], (short)0, acc[i][j], false, false);
  };

  // Triple-buffered pipeline: 2 stages of loads in flight; in-order counters
  // mean wait<=2 (TDM) / wait<=4 (async) == "stage ks has landed".
  const int NST = K_TOT / BK;  // 64 stages
  stage_load(0, 0);
  stage_load(1, 1);
  int buf = 0, nb = 2;
#pragma clang loop unroll(disable)
  for (int ks = 0; ks < NST - 2; ++ks) {
    wait_steady();         // stage ks data complete (ks+1 may still be in flight)
    __syncthreads();       // visible to all; all waves done reading buffer nb
    stage_load(ks + 2, nb);
    compute(buf);
    buf = (buf == 2) ? 0 : buf + 1;
    nb  = (nb  == 2) ? 0 : nb + 1;
  }
  wait_steady();           // stage NST-2 complete
  __syncthreads();
  compute(buf);
  buf = (buf == 2) ? 0 : buf + 1;
  wait_last();             // stage NST-1 complete
  __syncthreads();
  compute(buf);

  // epilogue: C layout — VGPR r: lanes0-15 M=r, lanes16-31 M=r+8; N=lane&15
#pragma unroll
  for (int i = 0; i < 4; ++i) {
#pragma unroll
    for (int j = 0; j < 2; ++j) {
      const int col = n0 + waveN * 32 + j * 16 + lh;
      const float bv = bias[col];
#pragma unroll
      for (int r = 0; r < 8; ++r) {
        const int rowg = m0 + waveM * 64 + i * 16 + r + hh * 8;
        C[(size_t)rowg * N_TOT + col] = acc[i][j][r] + bv;
      }
    }
  }
}

extern "C" void kernel_launch(void* const* d_in, const int* in_sizes, int n_in,
                              void* d_out, int out_size, void* d_ws, size_t ws_size,
                              hipStream_t stream) {
  (void)in_sizes; (void)n_in; (void)out_size; (void)ws_size;
  const float* x     = (const float*)d_in[0];
  const float* w     = (const float*)d_in[1];
  const float* bias  = (const float*)d_in[2];
  const float* gamma = (const float*)d_in[3];
  float* out = (float*)d_out;

  char* ws = (char*)d_ws;
  float*    thr      = (float*)ws;                 // 1 float
  float*    partials = (float*)(ws + 256);         // 1024 floats
  _Float16* xh       = (_Float16*)(ws + 8192);                                  // 64 MB
  _Float16* wh       = (_Float16*)(ws + 8192 + (size_t)M_TOT * K_TOT * 2);      // 32 MB

  absw_partial<<<1024, 256, 0, stream>>>(w, partials);
  absw_final<<<1, 256, 0, stream>>>(partials, thr);
  quant_w<<<4096, 256, 0, stream>>>(w, thr, wh);
  rmsnorm_f16<<<M_TOT, 256, 0, stream>>>(x, gamma, xh);

  dim3 grid(N_TOT / BN, M_TOT / BM);
  gemm_wmma<<<grid, 256, 0, stream>>>(xh, wh, bias, out);
}
